// MultiOmicsHeteroGNN_59768764891881
// MI455X (gfx1250) — matrix operand order repaired and checked by reading, not compiled, not run on hardware
//
#include <hip/hip_runtime.h>

#define HID 128
#define NEG_SLOPE 0.2f

typedef __attribute__((ext_vector_type(16))) _Float16 v16h;
typedef __attribute__((ext_vector_type(8)))  _Float16 v8h;
typedef __attribute__((ext_vector_type(8)))  float    v8f;

__device__ __forceinline__ int ordf(float f) {
    int i = __float_as_int(f);
    return i >= 0 ? i : (i ^ 0x7fffffff);
}
__device__ __forceinline__ float unordf(int k) {
    int u = k >= 0 ? k : (k ^ 0x7fffffff);
    return __int_as_float(u);
}

// ---------------- input embedding: h = relu(x * Wrow + b), x is [N,1] ----------------
__global__ void embed_kernel(const float* __restrict__ x, const float* __restrict__ W,
                             const float* __restrict__ b, float* __restrict__ h, int N) {
    int idx = blockIdx.x * blockDim.x + threadIdx.x;
    if (idx >= N * HID) return;
    int n = idx >> 7, c = idx & 127;
    float v = x[n] * W[c] + b[c];
    h[idx] = v > 0.f ? v : 0.f;
}

// ---------------- C[N,128] = A[N,128] @ W[128,128] (+bias), WMMA f16->f32 ----------------
__global__ __launch_bounds__(256) void gemm128_wmma(const float* __restrict__ A, int N,
                                                    const float* __restrict__ W,
                                                    const float* __restrict__ bias,
                                                    float* __restrict__ C) {
    __shared__ alignas(16) _Float16 As[128][136];   // [row][in-ch], padded rows (272B, 16B-aligned)
    __shared__ alignas(16) _Float16 Bt[128][136];   // [out-ch][in-ch] (W transposed)
    const int t = threadIdx.x;
    const int row0 = blockIdx.x * 128;

    for (int i = t; i < 128 * 128; i += 256) {      // stage W transposed as f16
        int k = i >> 7, o = i & 127;
        Bt[o][k] = (_Float16)W[i];
    }
    for (int i = t; i < 128 * 128; i += 256) {      // stage A tile as f16 (zero-pad tail rows)
        int r = i >> 7, c = i & 127;
        int gr = row0 + r;
        As[r][c] = (gr < N) ? (_Float16)A[(size_t)gr * HID + c] : (_Float16)0.f;
    }
    __syncthreads();

    const int wave = t >> 5, lane = t & 31;
    const int hi = lane >> 4;        // lane half: selects K sub-band per ISA layout
    const int l15 = lane & 15;
    const int arow = (wave << 4) | l15;

    v8f acc[8] = {};
    #pragma unroll
    for (int kk = 0; kk < 128; kk += 32) {
        // A 16x32 f16 fragment: lanes 0-15 K=kk+0..7,kk+16..23 ; lanes 16-31 K=kk+8..15,kk+24..31
        v8h alo = *(const v8h*)&As[arow][kk + hi * 8];
        v8h ahi = *(const v8h*)&As[arow][kk + 16 + hi * 8];
        v16h af = __builtin_shufflevector(alo, ahi, 0,1,2,3,4,5,6,7,8,9,10,11,12,13,14,15);
        const int kb = kk + hi * 8; // B 32x16: lanes 0-15 K=kk+0..7,kk+16..23 ; lanes 16-31 K=kk+8..15,kk+24..31
        #pragma unroll
        for (int s = 0; s < 8; s++) {
            int ocol = (s << 4) | l15;
            v8h blo = *(const v8h*)&Bt[ocol][kb];
            v8h bhi = *(const v8h*)&Bt[ocol][kb + 16];
            v16h bf = __builtin_shufflevector(blo, bhi, 0,1,2,3,4,5,6,7,8,9,10,11,12,13,14,15);
            acc[s] = __builtin_amdgcn_wmma_f32_16x16x32_f16(false, af, false, bf,
                                                            (short)0, acc[s], false, false);
        }
    }
    // C/D layout: VGPR r -> (M = r + 8*hi, N = l15 + 16*s)
    #pragma unroll
    for (int s = 0; s < 8; s++) {
        int col = (s << 4) | l15;
        float bv = bias ? bias[col] : 0.f;
        #pragma unroll
        for (int r = 0; r < 8; r++) {
            int row = row0 + (wave << 4) + (hi << 3) + r;
            if (row < N) C[(size_t)row * HID + col] = acc[s][r] + bv;
        }
    }
}

// ---------------- per-node attention halves: out[n][h] = sum_d m[n][h*32+d]*avec[h*32+d] ----------------
__global__ void adot_kernel(const float* __restrict__ m, const float* __restrict__ avec,
                            float* __restrict__ out, int N) {
    int wave = (blockIdx.x * blockDim.x + threadIdx.x) >> 5;
    int lane = threadIdx.x & 31;
    if (wave >= N) return;
    const float* row = m + (size_t)wave * HID;
    float s0 = row[lane]       * avec[lane];
    float s1 = row[32 + lane]  * avec[32 + lane];
    float s2 = row[64 + lane]  * avec[64 + lane];
    float s3 = row[96 + lane]  * avec[96 + lane];
    #pragma unroll
    for (int off = 16; off > 0; off >>= 1) {
        s0 += __shfl_xor(s0, off, 32);
        s1 += __shfl_xor(s1, off, 32);
        s2 += __shfl_xor(s2, off, 32);
        s3 += __shfl_xor(s3, off, 32);
    }
    if (lane == 0) {
        out[wave * 4 + 0] = s0; out[wave * 4 + 1] = s1;
        out[wave * 4 + 2] = s2; out[wave * 4 + 3] = s3;
    }
}

// ---------------- accumulator init with bias (optionally two biases summed) ----------------
__global__ void init_acc_kernel(float* __restrict__ acc, const float* __restrict__ b0,
                                const float* __restrict__ b1, int N) {
    int idx = blockIdx.x * blockDim.x + threadIdx.x;
    if (idx >= N * HID) return;
    int c = idx & 127;
    acc[idx] = b0[c] + (b1 ? b1[c] : 0.f);
}

__global__ void seginit_kernel(int* __restrict__ smax, float* __restrict__ den, int n4) {
    int idx = blockIdx.x * blockDim.x + threadIdx.x;
    if (idx >= n4) return;
    smax[idx] = ordf(-__builtin_huge_valf());
    den[idx] = 0.f;
}

// ---------------- pass 1: logits + segment max (ordered-int float atomicMax) ----------------
__global__ void edge_logit_kernel(const int* __restrict__ src, const int* __restrict__ dst,
                                  const float* __restrict__ aS, const float* __restrict__ aD,
                                  float* __restrict__ ebuf, int* __restrict__ smax, int E) {
    int t = blockIdx.x * blockDim.x + threadIdx.x;
    if (t >= E * 4) return;
    int e = t >> 2, h = t & 3;
    int s = src[e], d = dst[e];
    float x = aS[s * 4 + h] + aD[d * 4 + h];
    float lg = x > 0.f ? x : NEG_SLOPE * x;
    ebuf[t] = lg;
    atomicMax(&smax[d * 4 + h], ordf(lg));
}

// ---------------- pass 2: p = exp(logit - segmax), denom += p ----------------
__global__ void edge_exp_kernel(const int* __restrict__ dst, float* __restrict__ ebuf,
                                const int* __restrict__ smax, float* __restrict__ den, int E) {
    int t = blockIdx.x * blockDim.x + threadIdx.x;
    if (t >= E * 4) return;
    int e = t >> 2, h = t & 3;
    int d = dst[e];
    float sm = unordf(smax[d * 4 + h]);
    if (!isfinite(sm)) sm = 0.f;
    float p = __expf(ebuf[t] - sm);
    ebuf[t] = p;
    atomicAdd(&den[d * 4 + h], p);
}

// ---------------- pass 3: wave-per-edge weighted scatter of 128-float messages ----------------
__global__ __launch_bounds__(256) void scatter_kernel(const int* __restrict__ src,
                                                      const int* __restrict__ dst,
                                                      const float* __restrict__ m,
                                                      const float* __restrict__ ebuf,
                                                      const float* __restrict__ den,
                                                      float* __restrict__ acc, int E) {
    int e = (blockIdx.x * blockDim.x + threadIdx.x) >> 5;
    int lane = threadIdx.x & 31;
    if (e >= E) return;
    int s = src[e], d = dst[e];
    float aval = 0.f;
    if (lane < 4) {
        float dd = den[d * 4 + lane];
        aval = ebuf[e * 4 + lane] / fmaxf(dd, 1e-16f);
    }
    float a0 = __shfl(aval, 0, 32);
    float a1 = __shfl(aval, 1, 32);
    float a2 = __shfl(aval, 2, 32);
    float a3 = __shfl(aval, 3, 32);
    const float* mr = m + (size_t)s * HID;
    float* ar = acc + (size_t)d * HID;
    atomicAdd(&ar[lane],      a0 * mr[lane]);
    atomicAdd(&ar[32 + lane], a1 * mr[32 + lane]);
    atomicAdd(&ar[64 + lane], a2 * mr[64 + lane]);
    atomicAdd(&ar[96 + lane], a3 * mr[96 + lane]);
}

__global__ void relu_kernel(float* __restrict__ dstv, const float* __restrict__ srcv, int n) {
    int idx = blockIdx.x * blockDim.x + threadIdx.x;
    if (idx >= n) return;
    float v = srcv[idx];
    dstv[idx] = v > 0.f ? v : 0.f;
}

static inline int cdiv(int a, int b) { return (a + b - 1) / b; }

extern "C" void kernel_launch(void* const* d_in, const int* in_sizes, int n_in,
                              void* d_out, int out_size, void* d_ws, size_t ws_size,
                              hipStream_t stream) {
    (void)in_sizes; (void)n_in; (void)out_size; (void)ws_size;
    const float* x_g = (const float*)d_in[0];
    const float* x_p = (const float*)d_in[1];
    const float* x_m = (const float*)d_in[2];
    const int* gg_s = (const int*)d_in[3];
    const int* gg_d = (const int*)d_in[4];
    const int* gp_s = (const int*)d_in[5];
    const int* gp_d = (const int*)d_in[6];
    const int* pp_s = (const int*)d_in[7];
    const int* pp_d = (const int*)d_in[8];
    const int* pm_s = (const int*)d_in[9];
    const int* pm_d = (const int*)d_in[10];
    const float* W_emb = (const float*)d_in[11];
    const float* b_emb = (const float*)d_in[12];
    const float* W_conv = (const float*)d_in[13];
    const float* att_src = (const float*)d_in[14];
    const float* att_dst = (const float*)d_in[15];
    const float* b_conv = (const float*)d_in[16];
    const float* W_out = (const float*)d_in[17];
    const float* b_out = (const float*)d_in[18];
    float* out = (float*)d_out;

    const int NG = 50000, NP = 40000, NM = 20000;
    const int EGG = 800000, EGP = 400000, EPP = 600000, EPM = 200000;

    // workspace carve (all buffers fit well inside the 192MB L2 working-set regime)
    float* ws = (float*)d_ws;
    float* hg = ws;
    float* hp = hg + (size_t)NG * 128;
    float* hm = hp + (size_t)NP * 128;
    float* ag = hm + (size_t)NM * 128;
    float* ap = ag + (size_t)NG * 128;
    float* am = ap + (size_t)NP * 128;
    float* mA = am + (size_t)NM * 128;            // m of src type (max NG rows)
    float* mB = mA + (size_t)NG * 128;            // m of dst type (max NP rows)
    float* aS = mB + (size_t)NP * 128;            // [N,4]
    float* aD = aS + (size_t)NG * 4;
    int*   smax = (int*)(aD + (size_t)NG * 4);
    float* den = (float*)(smax + (size_t)NG * 4);
    float* ebuf = den + (size_t)NG * 4;           // per-edge per-head logits/numerators

    // input embeddings
    embed_kernel<<<cdiv(NG * 128, 256), 256, 0, stream>>>(x_g, W_emb + 0,   b_emb + 0,   hg, NG);
    embed_kernel<<<cdiv(NP * 128, 256), 256, 0, stream>>>(x_p, W_emb + 128, b_emb + 128, hp, NP);
    embed_kernel<<<cdiv(NM * 128, 256), 256, 0, stream>>>(x_m, W_emb + 256, b_emb + 256, hm, NM);

    for (int l = 0; l < 3; l++) {
        const float* Wl  = W_conv  + (size_t)l * 4 * 128 * 128;
        const float* asl = att_src + (size_t)l * 4 * 128;
        const float* adl = att_dst + (size_t)l * 4 * 128;
        const float* bl  = b_conv  + (size_t)l * 4 * 128;

        init_acc_kernel<<<cdiv(NG * 128, 256), 256, 0, stream>>>(ag, bl + 0 * 128, nullptr,    NG);
        init_acc_kernel<<<cdiv(NP * 128, 256), 256, 0, stream>>>(ap, bl + 1 * 128, bl + 2*128, NP);
        init_acc_kernel<<<cdiv(NM * 128, 256), 256, 0, stream>>>(am, bl + 3 * 128, nullptr,    NM);

        // edge type 0: g->g
        gemm128_wmma<<<cdiv(NG, 128), 256, 0, stream>>>(hg, NG, Wl + 0 * 16384, nullptr, mA);
        adot_kernel<<<cdiv(NG, 8), 256, 0, stream>>>(mA, asl + 0 * 128, aS, NG);
        adot_kernel<<<cdiv(NG, 8), 256, 0, stream>>>(mA, adl + 0 * 128, aD, NG);
        seginit_kernel<<<cdiv(NG * 4, 256), 256, 0, stream>>>(smax, den, NG * 4);
        edge_logit_kernel<<<cdiv(EGG * 4, 256), 256, 0, stream>>>(gg_s, gg_d, aS, aD, ebuf, smax, EGG);
        edge_exp_kernel<<<cdiv(EGG * 4, 256), 256, 0, stream>>>(gg_d, ebuf, smax, den, EGG);
        scatter_kernel<<<cdiv(EGG, 8), 256, 0, stream>>>(gg_s, gg_d, mA, ebuf, den, ag, EGG);

        // edge type 1: g->p
        gemm128_wmma<<<cdiv(NG, 128), 256, 0, stream>>>(hg, NG, Wl + 1 * 16384, nullptr, mA);
        gemm128_wmma<<<cdiv(NP, 128), 256, 0, stream>>>(hp, NP, Wl + 1 * 16384, nullptr, mB);
        adot_kernel<<<cdiv(NG, 8), 256, 0, stream>>>(mA, asl + 1 * 128, aS, NG);
        adot_kernel<<<cdiv(NP, 8), 256, 0, stream>>>(mB, adl + 1 * 128, aD, NP);
        seginit_kernel<<<cdiv(NP * 4, 256), 256, 0, stream>>>(smax, den, NP * 4);
        edge_logit_kernel<<<cdiv(EGP * 4, 256), 256, 0, stream>>>(gp_s, gp_d, aS, aD, ebuf, smax, EGP);
        edge_exp_kernel<<<cdiv(EGP * 4, 256), 256, 0, stream>>>(gp_d, ebuf, smax, den, EGP);
        scatter_kernel<<<cdiv(EGP, 8), 256, 0, stream>>>(gp_s, gp_d, mA, ebuf, den, ap, EGP);

        // edge type 2: p->p
        gemm128_wmma<<<cdiv(NP, 128), 256, 0, stream>>>(hp, NP, Wl + 2 * 16384, nullptr, mA);
        adot_kernel<<<cdiv(NP, 8), 256, 0, stream>>>(mA, asl + 2 * 128, aS, NP);
        adot_kernel<<<cdiv(NP, 8), 256, 0, stream>>>(mA, adl + 2 * 128, aD, NP);
        seginit_kernel<<<cdiv(NP * 4, 256), 256, 0, stream>>>(smax, den, NP * 4);
        edge_logit_kernel<<<cdiv(EPP * 4, 256), 256, 0, stream>>>(pp_s, pp_d, aS, aD, ebuf, smax, EPP);
        edge_exp_kernel<<<cdiv(EPP * 4, 256), 256, 0, stream>>>(pp_d, ebuf, smax, den, EPP);
        scatter_kernel<<<cdiv(EPP, 8), 256, 0, stream>>>(pp_s, pp_d, mA, ebuf, den, ap, EPP);

        // edge type 3: p->m
        gemm128_wmma<<<cdiv(NP, 128), 256, 0, stream>>>(hp, NP, Wl + 3 * 16384, nullptr, mA);
        gemm128_wmma<<<cdiv(NM, 128), 256, 0, stream>>>(hm, NM, Wl + 3 * 16384, nullptr, mB);
        adot_kernel<<<cdiv(NP, 8), 256, 0, stream>>>(mA, asl + 3 * 128, aS, NP);
        adot_kernel<<<cdiv(NM, 8), 256, 0, stream>>>(mB, adl + 3 * 128, aD, NM);
        seginit_kernel<<<cdiv(NM * 4, 256), 256, 0, stream>>>(smax, den, NM * 4);
        edge_logit_kernel<<<cdiv(EPM * 4, 256), 256, 0, stream>>>(pm_s, pm_d, aS, aD, ebuf, smax, EPM);
        edge_exp_kernel<<<cdiv(EPM * 4, 256), 256, 0, stream>>>(pm_d, ebuf, smax, den, EPM);
        scatter_kernel<<<cdiv(EPM, 8), 256, 0, stream>>>(pm_s, pm_d, mA, ebuf, den, am, EPM);

        // layer activation
        relu_kernel<<<cdiv(NG * 128, 256), 256, 0, stream>>>(hg, ag, NG * 128);
        relu_kernel<<<cdiv(NP * 128, 256), 256, 0, stream>>>(hp, ap, NP * 128);
        relu_kernel<<<cdiv(NM * 128, 256), 256, 0, stream>>>(hm, am, NM * 128);
    }

    // output projections (bias fused), written straight to d_out in (g,p,m) order
    gemm128_wmma<<<cdiv(NG, 128), 256, 0, stream>>>(hg, NG, W_out + 0 * 16384, b_out + 0,   out);
    gemm128_wmma<<<cdiv(NP, 128), 256, 0, stream>>>(hp, NP, W_out + 1 * 16384, b_out + 128, out + (size_t)NG * 128);
    gemm128_wmma<<<cdiv(NM, 128), 256, 0, stream>>>(hm, NM, W_out + 2 * 16384, b_out + 256, out + (size_t)(NG + NP) * 128);
}
